// AttnEncoder_19078244728975
// MI455X (gfx1250) — compile-verified, matrix operand order checked by reading, not run on hardware
//
#include <hip/hip_runtime.h>
#include <hip/hip_bf16.h>
#include <stddef.h>
#include <stdint.h>

// ---------------------------------------------------------------------------
// Problem sizes (fixed by the reference): B=256, T=256, N=512, H=512
// ---------------------------------------------------------------------------
#define BB 256
#define TT 256
#define NN 512
#define HH 512
#define GG (4 * HH)          // 2048 gate columns
#define KK (NN + HH)         // 1024 fused K (weighted | h)

typedef __attribute__((ext_vector_type(16))) __bf16          v16bf;
typedef __attribute__((ext_vector_type(8)))  float           v8f;
typedef __attribute__((ext_vector_type(8)))  unsigned short  ushort8;
typedef __attribute__((ext_vector_type(16))) unsigned short  ushort16;

union BfFrag { ushort16 u; v16bf v; ushort8 h[2]; };
union AccU  { v8f v; float f[8]; };

__device__ __forceinline__ unsigned short f2bf(float f) {
  unsigned int x = __float_as_uint(f);
  unsigned int r = x + 0x7FFFu + ((x >> 16) & 1u);   // round-to-nearest-even
  return (unsigned short)(r >> 16);
}

// CDNA5 async global->LDS copy (ASYNCcnt-tracked), 16B per lane.
__device__ __forceinline__ void async_b128(unsigned lds_off, const void* gaddr) {
  asm volatile("global_load_async_to_lds_b128 %0, %1, off"
               :: "v"(lds_off), "v"(gaddr) : "memory");
}

// ---------------------------------------------------------------------------
// One-time: pack fused weights W = [w_ih | w_hh] (2048 x 1024) transposed into
// WMMA B-fragment order for v_wmma_f32_16x16x32_bf16.
// Tile (kt, jt): 32 K x 16 N.  Per lane: 16 bf16 contiguous.
//   lane l<16  : n = jt*16+l,      elems 0..7 -> k = kt*32 + e,  8..15 -> +16
//   lane l>=16 : n = jt*16+(l-16), elems 0..7 -> k = kt*32+8+e,  8..15 -> +24
// ---------------------------------------------------------------------------
__global__ void pack_W(const float* __restrict__ w_ih,
                       const float* __restrict__ w_hh,
                       unsigned short* __restrict__ Wpk) {
  int gid  = blockIdx.x * blockDim.x + threadIdx.x;   // tile*32 + lane
  int lane = gid & 31;
  int tile = gid >> 5;                                // kt*128 + jt
  int jt   = tile & 127;
  int kt   = tile >> 7;
  int n    = jt * 16 + (lane & 15);
  int kadd = (lane < 16) ? 0 : 8;
  unsigned short* dst = Wpk + (size_t)gid * 16;
#pragma unroll
  for (int e = 0; e < 16; ++e) {
    int klocal = (e < 8) ? (kadd + e) : (16 + kadd + (e - 8));
    int k = kt * 32 + klocal;
    float v = (k < NN) ? w_ih[(size_t)n * NN + k]
                       : w_hh[(size_t)n * HH + (k - NN)];
    dst[e] = f2bf(v);
  }
}

// ---------------------------------------------------------------------------
// proj_x[b,n] = sum_t x[b,t,n] * w_x[t]; then row softmax -> A_soft[b,n].
// (h.w_h + c.w_c + attn_b are per-row constants => softmax-invariant, dropped.)
// ---------------------------------------------------------------------------
__global__ void proj_softmax(const float* __restrict__ x,
                             const float* __restrict__ attn_w,
                             float* __restrict__ A_soft) {
  __shared__ float wx[TT];
  __shared__ float red[NN];
  int b = blockIdx.x;
  int n = threadIdx.x;
  if (n < TT) wx[n] = attn_w[2 * HH + n];
  __syncthreads();

  float acc = 0.f;
  const float* xb = x + (size_t)b * TT * NN + n;
#pragma unroll 4
  for (int t = 0; t < TT; ++t) acc += xb[(size_t)t * NN] * wx[t];

  red[n] = acc;
  __syncthreads();
  for (int s = NN / 2; s > 0; s >>= 1) {
    if (n < s) red[n] = fmaxf(red[n], red[n + s]);
    __syncthreads();
  }
  float m = red[0];
  __syncthreads();
  float e = __expf(acc - m);
  red[n] = e;
  __syncthreads();
  for (int s = NN / 2; s > 0; s >>= 1) {
    if (n < s) red[n] += red[n + s];
    __syncthreads();
  }
  A_soft[(size_t)b * NN + n] = e / red[0];
}

// ---------------------------------------------------------------------------
// attentions[b,t,n] = A_soft[b,n] for all t (a_t is t-invariant).
// ---------------------------------------------------------------------------
__global__ void bcast_attn(const float* __restrict__ A_soft,
                           float* __restrict__ out) {
  size_t i  = (size_t)blockIdx.x * blockDim.x + threadIdx.x; // float4 index
  int    n4 = (int)(i & (NN / 4 - 1));                       // 0..127
  size_t bt = i >> 7;
  int    b  = (int)(bt >> 8);                                // /T
  float4 v  = ((const float4*)(A_soft + (size_t)b * NN))[n4];
  ((float4*)out)[i] = v;
}

// ---------------------------------------------------------------------------
// State init: c_ws = c0 (f32), h_bf = bf16(h0).
// ---------------------------------------------------------------------------
__global__ void init_state(const float* __restrict__ h0,
                           const float* __restrict__ c0,
                           unsigned short* __restrict__ hbf,
                           float* __restrict__ c) {
  int i = blockIdx.x * blockDim.x + threadIdx.x;  // B*H = 131072
  c[i]   = c0[i];
  hbf[i] = f2bf(h0[i]);
}

// ---------------------------------------------------------------------------
// weighted_bf16[b,n] = bf16(A_soft[b,n] * x[b,t,n])
// ---------------------------------------------------------------------------
__global__ void weighted_k(const float* __restrict__ A_soft,
                           const float* __restrict__ x, int t,
                           unsigned short* __restrict__ wbf) {
  int i = blockIdx.x * blockDim.x + threadIdx.x;  // B*N = 131072
  int b = i >> 9;
  int n = i & (NN - 1);
  float v = A_soft[i] * x[((size_t)(b * TT + t)) * NN + n];
  wbf[i] = f2bf(v);
}

// ---------------------------------------------------------------------------
// gates = [weighted | h] (256x1024, bf16) x Wpk (1024x2048).
// Block = 128 threads (4 waves) -> 64(M) x 64(N) tile.
//   wave w owns M-tile (blockIdx.y*4 + w) and all 4 N-tiles of blockIdx.x.
// B fragments (shared by the 4 waves) staged into LDS with
// global_load_async_to_lds_b128, double-buffered (ASYNCcnt FIFO pipeline).
// K loop unrolled x2 with compile-time buffer parity and two alternating
// A-fragment registers -> no loop-carried register copies.
// ---------------------------------------------------------------------------
__global__ void __launch_bounds__(128)
gemm_gates(const unsigned short* __restrict__ Wpk,
           const unsigned short* __restrict__ Aw,   // weighted bf16, 256x512
           const unsigned short* __restrict__ Ah,   // h bf16, 256x512
           float* __restrict__ gates) {
  // smem: [2 buffers][4 N-tiles][32 lanes][32 bytes] = 8 KB
  __shared__ __align__(16) unsigned short smem[2 * 4 * 32 * 16];

  const int lane = threadIdx.x & 31;
  const int wave = threadIdx.x >> 5;          // 0..3  (also: which B tile I stage)
  const int ng   = blockIdx.x;                // 0..31 (group of 4 N tiles)
  const int mt   = blockIdx.y * 4 + wave;     // 0..15 (my M tile)

  const int row  = mt * 16 + (lane & 15);
  const int koff = (lane < 16) ? 0 : 8;

  const unsigned lds0   = (unsigned)(uintptr_t)smem;                // LDS byte offset
  const unsigned dstoff = (unsigned)((wave * 32 + lane) * 32);      // within a buffer

  v8f acc0 = {}, acc1 = {}, acc2 = {}, acc3 = {};

  // B source for (kt, tile = ng*4+wave): byte stride per kt = 128*32*32 = 128 KB
  const char* wsrc =
      (const char*)Wpk + ((size_t)((ng * 4 + wave) * 32 + lane)) * 32;

  auto issueB = [&](int kt, int buf) {
    const char* src = wsrc + (size_t)kt * (128 * 32 * 32);
    unsigned d = lds0 + (unsigned)(buf * 4096) + dstoff;
    async_b128(d, src);
    async_b128(d + 16, src + 16);
  };
  auto loadA = [&](int kt) {
    const unsigned short* Asec = (kt < 16) ? Aw : Ah;
    const int kcol = ((kt & 15) * 32) + koff;
    BfFrag a;
    a.h[0] = *(const ushort8*)(Asec + (size_t)row * NN + kcol);
    a.h[1] = *(const ushort8*)(Asec + (size_t)row * NN + kcol + 16);
    return a;
  };
  auto compute = [&](int buf, const BfFrag& a) {
    const unsigned short* bbase = smem + buf * 2048;  // shorts; buf is literal
    BfFrag b0, b1, b2, b3;
    b0.h[0] = *(const ushort8*)(bbase + (0 * 32 + lane) * 16);
    b0.h[1] = *(const ushort8*)(bbase + (0 * 32 + lane) * 16 + 8);
    b1.h[0] = *(const ushort8*)(bbase + (1 * 32 + lane) * 16);
    b1.h[1] = *(const ushort8*)(bbase + (1 * 32 + lane) * 16 + 8);
    b2.h[0] = *(const ushort8*)(bbase + (2 * 32 + lane) * 16);
    b2.h[1] = *(const ushort8*)(bbase + (2 * 32 + lane) * 16 + 8);
    b3.h[0] = *(const ushort8*)(bbase + (3 * 32 + lane) * 16);
    b3.h[1] = *(const ushort8*)(bbase + (3 * 32 + lane) * 16 + 8);
    acc0 = __builtin_amdgcn_wmma_f32_16x16x32_bf16(false, a.v, false, b0.v,
                                                   (short)0, acc0, false, false);
    acc1 = __builtin_amdgcn_wmma_f32_16x16x32_bf16(false, a.v, false, b1.v,
                                                   (short)0, acc1, false, false);
    acc2 = __builtin_amdgcn_wmma_f32_16x16x32_bf16(false, a.v, false, b2.v,
                                                   (short)0, acc2, false, false);
    acc3 = __builtin_amdgcn_wmma_f32_16x16x32_bf16(false, a.v, false, b3.v,
                                                   (short)0, acc3, false, false);
  };

  // Prime both buffers (4 asyncs in flight; ASYNCcnt is FIFO).
  issueB(0, 0);
  BfFrag a0 = loadA(0);
  issueB(1, 1);
  BfFrag a1 = loadA(1);

  for (int kt = 0; kt < 32; kt += 2) {
    // ---- even step: buffer 0 ----
    asm volatile("s_wait_asynccnt 0x2" ::: "memory");  // buf0's copies landed
    __syncthreads();
    compute(0, a0);
    __syncthreads();                                   // all waves done with buf0
    if (kt + 2 < 32) {
      issueB(kt + 2, 0);                               // refill buf0
      a0 = loadA(kt + 2);                              // register prefetch of A
      asm volatile("s_wait_asynccnt 0x2" ::: "memory");// buf1's copies landed
    } else {
      asm volatile("s_wait_asynccnt 0x0" ::: "memory");
    }
    // ---- odd step: buffer 1 ----
    __syncthreads();
    compute(1, a1);
    __syncthreads();                                   // all waves done with buf1
    if (kt + 3 < 32) {
      issueB(kt + 3, 1);                               // refill buf1
      a1 = loadA(kt + 3);
    }
  }

  // ---- store: C/D layout: VGPR r, lanes<16 -> M=r, lanes>=16 -> M=r+8 ----
  const int rbase = mt * 16 + ((lane < 16) ? 0 : 8);
  const int cl    = lane & 15;
  AccU u0, u1, u2, u3;
  u0.v = acc0; u1.v = acc1; u2.v = acc2; u3.v = acc3;
#pragma unroll
  for (int r = 0; r < 8; ++r) {
    float* g = gates + (size_t)(rbase + r) * GG + (ng * 4) * 16 + cl;
    g[0]  = u0.f[r];
    g[16] = u1.f[r];
    g[32] = u2.f[r];
    g[48] = u3.f[r];
  }
}

// ---------------------------------------------------------------------------
// LSTM pointwise cell: bias add, gate nonlinearities, state update,
// writes c (f32), h (bf16 for next GEMM), and input_encoded[:, t, :].
// ---------------------------------------------------------------------------
__global__ void lstm_cell(const float* __restrict__ gates,
                          const float* __restrict__ b_ih,
                          const float* __restrict__ b_hh,
                          float* __restrict__ c,
                          unsigned short* __restrict__ hbf,
                          float* __restrict__ enc_out, int t) {
  int b = blockIdx.x;
  int j = threadIdx.x;
  size_t base = (size_t)b * GG;
  float gi = gates[base + j]            + b_ih[j]            + b_hh[j];
  float gf = gates[base + HH + j]       + b_ih[HH + j]       + b_hh[HH + j];
  float gg = gates[base + 2 * HH + j]   + b_ih[2 * HH + j]   + b_hh[2 * HH + j];
  float go = gates[base + 3 * HH + j]   + b_ih[3 * HH + j]   + b_hh[3 * HH + j];

  float is = 1.f / (1.f + __expf(-gi));
  float fs = 1.f / (1.f + __expf(-gf));
  float gt = tanhf(gg);
  float os = 1.f / (1.f + __expf(-go));

  size_t si = (size_t)b * HH + j;
  float cn = fs * c[si] + is * gt;
  float hn = os * tanhf(cn);
  c[si]   = cn;
  hbf[si] = f2bf(hn);
  enc_out[((size_t)(b * TT + t)) * HH + j] = hn;
}

// ---------------------------------------------------------------------------
// Host-side orchestration (graph-capturable: only kernel launches on stream).
// d_in order: input_data, h0, c0, attn_w, attn_b, w_ih, w_hh, b_ih, b_hh
// d_out: attentions (B*T*N floats) then input_encoded (B*T*H floats)
// ---------------------------------------------------------------------------
extern "C" void kernel_launch(void* const* d_in, const int* in_sizes, int n_in,
                              void* d_out, int out_size, void* d_ws, size_t ws_size,
                              hipStream_t stream) {
  const float* x      = (const float*)d_in[0];
  const float* h0     = (const float*)d_in[1];
  const float* c0     = (const float*)d_in[2];
  const float* attn_w = (const float*)d_in[3];
  // d_in[4] = attn_b: per-row constant, softmax-invariant -> unused
  const float* w_ih   = (const float*)d_in[5];
  const float* w_hh   = (const float*)d_in[6];
  const float* b_ih   = (const float*)d_in[7];
  const float* b_hh   = (const float*)d_in[8];

  float* attn_out = (float*)d_out;                              // B*T*N
  float* enc_out  = (float*)d_out + (size_t)BB * TT * NN;       // B*T*H

  // Workspace layout (bytes), all offsets 256B-aligned:
  char* ws = (char*)d_ws;
  unsigned short* Wpk    = (unsigned short*)(ws);                 // 4 MB  (2048x1024 bf16 packed)
  float*          A_soft = (float*)(ws + 4194304);                // 512 KB (B x N)
  unsigned short* wbf    = (unsigned short*)(ws + 4718592);       // 256 KB (B x N bf16)
  unsigned short* hbf    = (unsigned short*)(ws + 4980736);       // 256 KB (B x H bf16)
  float*          c_ws   = (float*)(ws + 5242880);                // 512 KB (B x H)
  float*          gates  = (float*)(ws + 5767168);                // 2 MB  (B x 4H)
  // total: 7,864,320 bytes

  // One-time setup
  pack_W      <<<512,   256, 0, stream>>>(w_ih, w_hh, Wpk);
  proj_softmax<<<BB,    NN,  0, stream>>>(x, attn_w, A_soft);
  bcast_attn  <<<32768, 256, 0, stream>>>(A_soft, attn_out);
  init_state  <<<512,   256, 0, stream>>>(h0, c0, hbf, c_ws);

  // Sequential scan over T
  dim3 ggrid(32, 4);
  for (int t = 0; t < TT; ++t) {
    weighted_k<<<512, 256, 0, stream>>>(A_soft, x, t, wbf);
    gemm_gates<<<ggrid, 128, 0, stream>>>(Wpk, wbf, hbf, gates);
    lstm_cell <<<BB,  HH,  0, stream>>>(gates, b_ih, b_hh, c_ws, hbf, enc_out, t);
  }
}
